// ASTAttention_11269994185376
// MI455X (gfx1250) — matrix operand order repaired and checked by reading, not compiled, not run on hardware
//
#include <hip/hip_runtime.h>
#include <hip/hip_bf16.h>
#include <math.h>
#include <stdint.h>

#define BB 2
#define NN 2048
#define DD 768
#define HH 12
#define HDIM 64
#define LDST 66   // padded LDS row stride (elements)

typedef __attribute__((ext_vector_type(16))) __bf16        v16bf;
typedef __attribute__((ext_vector_type(8)))  float         v8f;
typedef __attribute__((ext_vector_type(8)))  unsigned int  v8u;

// ---------- bf16 helpers (round-to-nearest-even) ----------
__device__ __forceinline__ unsigned short f2bf(float f) {
  unsigned u = __builtin_bit_cast(unsigned, f);
  u += 0x7FFFu + ((u >> 16) & 1u);
  return (unsigned short)(u >> 16);
}
__device__ __forceinline__ unsigned pack2(float a, float b) {
  return (unsigned)f2bf(a) | ((unsigned)f2bf(b) << 16);
}

// ---------- WMMA wrapper ----------
__device__ __forceinline__ v8f wmma_bf16(v16bf a, v16bf b, v8f c) {
  return __builtin_amdgcn_wmma_f32_16x16x32_bf16(false, a, false, b,
                                                 (short)0, c, false, false);
}

// ---------- fragment builders from bf16 LDS tiles ----------
// A-matrix 16x32 (ISA 7.12.2): lane m = lane&15, g = lane>>4.
// VGPR j<4: K = 2j + 8g (+pair); j>=4: K = 8 + 2j + 8g (+pair).
__device__ __forceinline__ v16bf load_afrag(const unsigned short* lds, int row,
                                            int kc, int g) {
  v8u t;
  const int base = row * LDST + kc + 8 * g;
#pragma unroll
  for (int j = 0; j < 8; ++j) {
    const int kb = (j < 4) ? (2 * j) : (8 + 2 * j);
    t[j] = *reinterpret_cast<const unsigned*>(lds + base + kb);
  }
  return __builtin_bit_cast(v16bf, t);
}
// B-matrix 32x16: lane n = lane&15, g = lane>>4; same K order as A:
// VGPR j<4: K = 2j + 8g (+pair); j>=4: K = 8 + 2j + 8g (+pair). LDS tile indexed [n][k].
__device__ __forceinline__ v16bf load_bfrag(const unsigned short* lds, int col,
                                            int kc, int g) {
  v8u t;
  const int base = col * LDST + kc + 8 * g;
#pragma unroll
  for (int j = 0; j < 8; ++j) {
    const int kb = (j < 4) ? (2 * j) : (8 + 2 * j);
    t[j] = *reinterpret_cast<const unsigned*>(lds + base + kb);
  }
  return __builtin_bit_cast(v16bf, t);
}

// ---------- cooperative 64x64 f32 tile -> bf16 LDS staging (128 threads) ----
__device__ __forceinline__ void stage_tile(const float* __restrict__ gp, int ld,
                                           unsigned short* lds, int tid) {
#pragma unroll
  for (int p = 0; p < 8; ++p) {
    const int r = p * 8 + (tid >> 4);
    const int c = (tid & 15) * 4;
    float4 v = *reinterpret_cast<const float4*>(gp + (size_t)r * ld + c);
    unsigned* dst = reinterpret_cast<unsigned*>(lds + r * LDST + c);
    dst[0] = pack2(v.x, v.y);
    dst[1] = pack2(v.z, v.w);
  }
}
// transposed: tile element (r,c) -> lds[c][r]
__device__ __forceinline__ void stage_tile_tr(const float* __restrict__ gp, int ld,
                                              unsigned short* lds, int tid) {
#pragma unroll
  for (int p = 0; p < 8; ++p) {
    const int r = p * 8 + (tid >> 4);
    const int c = (tid & 15) * 4;
    float4 v = *reinterpret_cast<const float4*>(gp + (size_t)r * ld + c);
    lds[(c + 0) * LDST + r] = f2bf(v.x);
    lds[(c + 1) * LDST + r] = f2bf(v.y);
    lds[(c + 2) * LDST + r] = f2bf(v.z);
    lds[(c + 3) * LDST + r] = f2bf(v.w);
  }
}

// ============================================================
// GEMM: C[M,Nc] = A[M,K] @ W[K,Nc] + bias[Nc]   (bf16 WMMA, f32 accum)
// grid = (Nc/64, M/64), block = 128 (4 waves)
// ============================================================
__global__ __launch_bounds__(128) void gemm_bias(const float* __restrict__ A,
                                                 const float* __restrict__ W,
                                                 const float* __restrict__ bias,
                                                 float* __restrict__ C,
                                                 int K, int Nc) {
  __shared__ unsigned short At[64 * LDST];   // [row][k]
  __shared__ unsigned short Wt[64 * LDST];   // [n][k]  (transposed)
  const int tid = threadIdx.x;
  const int w = tid >> 5, l = tid & 31, g = l >> 4, ln = l & 15;
  const int col0 = blockIdx.x * 64;
  const int row0 = blockIdx.y * 64;

  v8f acc[4] = {};
  for (int kc = 0; kc < K; kc += 64) {
    __syncthreads();
    stage_tile(A + (size_t)row0 * K + kc, K, At, tid);
    stage_tile_tr(W + (size_t)kc * Nc + col0, Nc, Wt, tid);
    if (kc + 64 < K) {   // warm L2/L0 for the next K-chunk
      const int r = tid & 63;
      __builtin_prefetch(A + (size_t)(row0 + r) * K + kc + 64, 0, 1);
      __builtin_prefetch(W + (size_t)(kc + 64 + r) * Nc + col0, 0, 1);
    }
    __syncthreads();
    const v16bf a0 = load_afrag(At, 16 * w + ln, 0, g);
    const v16bf a1 = load_afrag(At, 16 * w + ln, 32, g);
#pragma unroll
    for (int ns = 0; ns < 4; ++ns) {
      const v16bf b0 = load_bfrag(Wt, 16 * ns + ln, 0, g);
      const v16bf b1 = load_bfrag(Wt, 16 * ns + ln, 32, g);
      acc[ns] = wmma_bf16(a0, b0, acc[ns]);
      acc[ns] = wmma_bf16(a1, b1, acc[ns]);
    }
  }
#pragma unroll
  for (int ns = 0; ns < 4; ++ns) {
    const int col = col0 + 16 * ns + ln;
    const float bv = bias[col];
#pragma unroll
    for (int r = 0; r < 8; ++r) {
      const int row = row0 + 16 * w + r + 8 * g;
      C[(size_t)row * Nc + col] = acc[ns][r] + bv;
    }
  }
}

// ============================================================
// Flash attention with relative-position bias + key mask.
// grid = (N/64, H, B), block = 128 (4 waves); q-tile = 64 rows.
// qkv layout: [B*N][3D]; Q at col h*64, K at D+h*64, V at 2D+h*64.
// dist tile is streamed into LDS with global_load_async_to_lds_b128
// (ASYNCcnt) so the QK^T WMMA phase hides its latency.
// ============================================================
__global__ __launch_bounds__(128) void attn_kernel(const float* __restrict__ qkv,
                                                   const int* __restrict__ dist,
                                                   const int* __restrict__ mask,
                                                   const float* __restrict__ bias_table,
                                                   float* __restrict__ obuf) {
  __shared__ unsigned short Kt[64 * LDST];        // [key][hd]
  __shared__ unsigned short Vt[64 * LDST];        // [hd][key] (transposed)
  __shared__ unsigned short Pt[64 * LDST];        // [row][key] (also Q staging)
  __shared__ float St[64 * LDST];                 // scores f32
  __shared__ __align__(16) int Dt[64 * 64];       // dist tile (async-loaded)
  __shared__ float pmax[128], psum[128];
  __shared__ float mrow[64], lrow[64], arow[64];
  __shared__ int   mtile[64];
  __shared__ float btab[32];

  const int tid = threadIdx.x;
  const int w = tid >> 5, l = tid & 31, g = l >> 4, ln = l & 15;
  const int qt = blockIdx.x, h = blockIdx.y, b = blockIdx.z;
  const int q0 = qt * 64;
  const size_t rowbase = (size_t)(b * NN + q0);

  // ---- stage Q tile, per-head bias column, init row stats
  stage_tile(qkv + rowbase * (3 * DD) + h * HDIM, 3 * DD, Pt, tid);
  if (tid < 64) { mrow[tid] = -INFINITY; lrow[tid] = 0.0f; }
  if (tid < 32) btab[tid] = bias_table[tid * HH + h];
  __syncthreads();
  const v16bf aq0 = load_afrag(Pt, 16 * w + ln, 0, g);
  const v16bf aq1 = load_afrag(Pt, 16 * w + ln, 32, g);

  v8f oacc[4] = {};

  for (int kv = 0; kv < NN / 64; ++kv) {
    const int key0 = kv * 64;
    __syncthreads();   // previous iteration done with Kt/Vt/Pt/Dt

    // ---- async: dist[q0..q0+63][key0..key0+63] -> LDS (ASYNCcnt)
#pragma unroll
    for (int p = 0; p < 8; ++p) {
      const int r = p * 8 + (tid >> 4);
      const int c = (tid & 15) * 4;
      const int* ga = dist + ((size_t)(b * NN + q0 + r)) * NN + key0 + c;
      const unsigned ldsoff = (unsigned)(uintptr_t)(Dt + r * 64 + c);
      asm volatile("global_load_async_to_lds_b128 %0, %1, off"
                   :: "v"(ldsoff), "v"(ga) : "memory");
    }

    // ---- stage K (row-major) and V (transposed) as bf16
    stage_tile(qkv + (size_t)(b * NN + key0) * (3 * DD) + DD + h * HDIM,
               3 * DD, Kt, tid);
    stage_tile_tr(qkv + (size_t)(b * NN + key0) * (3 * DD) + 2 * DD + h * HDIM,
                  3 * DD, Vt, tid);
    if (tid < 64) mtile[tid] = mask[(size_t)b * NN + key0 + tid];

    // ---- prefetch next kv tile's K/V rows (one line per thread)
    if (kv + 1 < NN / 64) {
      const int r = tid & 63, which = tid >> 6;   // 0 -> K, 1 -> V
      __builtin_prefetch(qkv + (size_t)(b * NN + key0 + 64 + r) * (3 * DD) +
                             (1 + which) * DD + h * HDIM, 0, 1);
    }
    __syncthreads();

    // ---- S = Q . K^T  (B-matrix element [kd][n] = Kt[n][kd])
#pragma unroll
    for (int ns = 0; ns < 4; ++ns) {
      const v16bf bk0 = load_bfrag(Kt, 16 * ns + ln, 0, g);
      const v16bf bk1 = load_bfrag(Kt, 16 * ns + ln, 32, g);
      v8f s = {};
      s = wmma_bf16(aq0, bk0, s);
      s = wmma_bf16(aq1, bk1, s);
#pragma unroll
      for (int r = 0; r < 8; ++r)
        St[(16 * w + r + 8 * g) * LDST + 16 * ns + ln] = s[r];
    }
    asm volatile("s_wait_asynccnt 0x0" ::: "memory");  // Dt resident (this wave)
    __syncthreads();                                   // ...and all waves

    // ---- softmax phase A: scale + bias + mask, partial row max
    {
      const int r = tid >> 1, cb = (tid & 1) * 32;
      float pm = -INFINITY;
      for (int c = cb; c < cb + 32; ++c) {
        float s = St[r * LDST + c] * 0.125f;   // 1/sqrt(64)
        int d = Dt[r * 64 + c];
        d = (d < 0) ? 0 : ((d > 31) ? 31 : d);
        s += btab[d];
        if (mtile[c]) s = -INFINITY;
        St[r * LDST + c] = s;
        pm = fmaxf(pm, s);
      }
      pmax[tid] = pm;
    }
    __syncthreads();

    // ---- softmax phase B: exp + partial sums, write P (bf16)
    {
      const int r = tid >> 1, cb = (tid & 1) * 32;
      const float nm = fmaxf(mrow[r], fmaxf(pmax[2 * r], pmax[2 * r + 1]));
      float sum = 0.0f;
      for (int c = cb; c < cb + 32; ++c) {
        const float s = St[r * LDST + c];
        const float p = (s == -INFINITY) ? 0.0f : __expf(s - nm);
        sum += p;
        Pt[r * LDST + c] = f2bf(p);
      }
      psum[tid] = sum;
    }
    __syncthreads();

    // ---- softmax phase C: fold running stats
    if (tid < 64) {
      const int r = tid;
      const float mold = mrow[r];
      const float nm = fmaxf(mold, fmaxf(pmax[2 * r], pmax[2 * r + 1]));
      const float al = (mold == -INFINITY) ? 0.0f : __expf(mold - nm);
      lrow[r] = al * lrow[r] + psum[2 * r] + psum[2 * r + 1];
      mrow[r] = nm;
      arow[r] = al;
    }
    __syncthreads();

    // ---- rescale accumulators, O += P . V
    float al[8];
#pragma unroll
    for (int r = 0; r < 8; ++r) al[r] = arow[16 * w + r + 8 * g];
    const v16bf ap0 = load_afrag(Pt, 16 * w + ln, 0, g);
    const v16bf ap1 = load_afrag(Pt, 16 * w + ln, 32, g);
#pragma unroll
    for (int ns = 0; ns < 4; ++ns) {
#pragma unroll
      for (int r = 0; r < 8; ++r) oacc[ns][r] *= al[r];
      const v16bf bv0 = load_bfrag(Vt, 16 * ns + ln, 0, g);
      const v16bf bv1 = load_bfrag(Vt, 16 * ns + ln, 32, g);
      oacc[ns] = wmma_bf16(ap0, bv0, oacc[ns]);
      oacc[ns] = wmma_bf16(ap1, bv1, oacc[ns]);
    }
  }

  // ---- normalize and write O  (obuf layout [B*N][D], head block h*64)
  float linv[8];
#pragma unroll
  for (int r = 0; r < 8; ++r) {
    const float lv = lrow[16 * w + r + 8 * g];
    linv[r] = (lv > 0.0f) ? (1.0f / lv) : 0.0f;
  }
#pragma unroll
  for (int ns = 0; ns < 4; ++ns) {
#pragma unroll
    for (int r = 0; r < 8; ++r) {
      const size_t row = rowbase + 16 * w + r + 8 * g;
      obuf[row * DD + h * HDIM + 16 * ns + ln] = oacc[ns][r] * linv[r];
    }
  }
}

// ============================================================
extern "C" void kernel_launch(void* const* d_in, const int* in_sizes, int n_in,
                              void* d_out, int out_size, void* d_ws, size_t ws_size,
                              hipStream_t stream) {
  const float* x          = (const float*)d_in[0];
  const int*   dist       = (const int*)d_in[1];
  const int*   mask       = (const int*)d_in[2];
  const float* qkv_w      = (const float*)d_in[3];
  const float* qkv_b      = (const float*)d_in[4];
  const float* out_w      = (const float*)d_in[5];
  const float* out_b      = (const float*)d_in[6];
  const float* bias_table = (const float*)d_in[7];
  float* out = (float*)d_out;

  float* qkv  = (float*)d_ws;                         // [B*N][3D]  37.7 MB
  float* obuf = qkv + (size_t)BB * NN * 3 * DD;       // [B*N][D]   12.6 MB

  // 1) qkv = x @ qkv_w + qkv_b
  gemm_bias<<<dim3((3 * DD) / 64, (BB * NN) / 64), 128, 0, stream>>>(
      x, qkv_w, qkv_b, qkv, DD, 3 * DD);
  // 2) flash attention with bias + mask
  attn_kernel<<<dim3(NN / 64, HH, BB), 128, 0, stream>>>(
      qkv, dist, mask, bias_table, obuf);
  // 3) out = o @ out_w + out_b
  gemm_bias<<<dim3(DD / 64, (BB * NN) / 64), 128, 0, stream>>>(
      obuf, out_w, out_b, out, DD, DD);
}
